// Attention_18923625906345
// MI455X (gfx1250) — matrix operand order repaired and checked by reading, not compiled, not run on hardware
//
#include <hip/hip_runtime.h>

// ---------------------------------------------------------------------------
// FlashAttention-2 forward, B=4,H=16,S=2048,D=64 fp32 (reference's causal
// branch is a no-op). Workgroup = 8 wave32 = 128 q rows; each wave owns a
// 16-row Q tile. KV visited in cooperative blocks of 64:
//   - block stages K (row-major bf16) and V (transposed bf16, Vt[d][kv]) in
//     LDS once per iteration (coalesced b128 loads, v_cvt_pk_bf16_f32),
//     amortizing loads+converts across all 8 waves
//   - per wave, every WMMA B operand is then two contiguous ds_load_b128
// bf16 WMMA (v_wmma_f32_16x16x32_bf16), fp32 accumulation: 8 WMMA (S=QK^T)
// + 2 WMMA (row sums via all-ones B tile) + 8 WMMA (O += P V) per iteration.
// Online softmax: row max via v_permlane16_b32 xor-butterfly; row sums ride
// the same alpha rescaling through the ones-tile WMMA; P transposed
// C-layout -> A-layout via a private per-wave LDS region. log2(e) folded
// into the scale so p = exp2(s-m) is one v_exp_f32.
// ---------------------------------------------------------------------------

typedef __attribute__((ext_vector_type(16))) __bf16 bf16x16;
typedef __attribute__((ext_vector_type(8)))  __bf16 bf16x8;
typedef __attribute__((ext_vector_type(8)))  float  f32x8;

#define SEQ     2048
#define DH      64
#define BM      16      // q rows per wave
#define BN      64      // kv columns per cooperative block
#define WAVES   8       // waves per workgroup -> 128 q rows per block
#define PITCH   72      // padded LDS row length (bf16): 144B rows, 16B align

#if __has_builtin(__builtin_amdgcn_permlane16)
#define USE_PERMLANE 1
#else
#define USE_PERMLANE 0
#endif

static __device__ __forceinline__ __bf16 f2bf(float f) { return (__bf16)f; }

static __device__ __forceinline__ float fast_exp2(float x) {
#if __has_builtin(__builtin_amdgcn_exp2f)
  return __builtin_amdgcn_exp2f(x);
#else
  return exp2f(x);
#endif
}

static __device__ __forceinline__ float fast_rcp(float x) {
#if __has_builtin(__builtin_amdgcn_rcpf)
  return __builtin_amdgcn_rcpf(x);
#else
  return 1.0f / x;
#endif
}

// max-reduce across the 16 lanes holding one C-matrix row (xor butterfly)
static __device__ __forceinline__ float rmax16(float x) {
#if USE_PERMLANE
  unsigned u, t;
  u = __builtin_bit_cast(unsigned, x);
  t = __builtin_amdgcn_permlane16(u, u, 0x67452301u, 0xEFCDAB89u, true, false); // xor 1
  x = fmaxf(x, __builtin_bit_cast(float, t));
  u = __builtin_bit_cast(unsigned, x);
  t = __builtin_amdgcn_permlane16(u, u, 0x54761032u, 0xDCFE98BAu, true, false); // xor 2
  x = fmaxf(x, __builtin_bit_cast(float, t));
  u = __builtin_bit_cast(unsigned, x);
  t = __builtin_amdgcn_permlane16(u, u, 0x32107654u, 0xBA98FEDCu, true, false); // xor 4
  x = fmaxf(x, __builtin_bit_cast(float, t));
  u = __builtin_bit_cast(unsigned, x);
  t = __builtin_amdgcn_permlane16(u, u, 0xFEDCBA98u, 0x76543210u, true, false); // xor 8
  x = fmaxf(x, __builtin_bit_cast(float, t));
  return x;
#else
  x = fmaxf(x, __shfl_xor(x, 1, 32));
  x = fmaxf(x, __shfl_xor(x, 2, 32));
  x = fmaxf(x, __shfl_xor(x, 4, 32));
  x = fmaxf(x, __shfl_xor(x, 8, 32));
  return x;
#endif
}

__global__ __launch_bounds__(WAVES * 32)
void fa_fwd_bf16wmma(const float* __restrict__ Q,
                     const float* __restrict__ K,
                     const float* __restrict__ V,
                     const float* __restrict__ scale_p,
                     float* __restrict__ O) {
  __shared__ __bf16 LdsK [BN][PITCH];        // K block, row-major  [kv][d]
  __shared__ __bf16 LdsVt[DH][PITCH];        // V block, transposed [d][kv]
  __shared__ __bf16 Pl[WAVES][BM][PITCH];    // per-wave P staging (transpose)

  const int tid  = threadIdx.x;
  const int lane = tid & 31;
  const int wave = tid >> 5;
  const int n    = lane & 15;      // column index within 16-wide tile
  const int h    = lane >> 4;      // half-wave select

  const int qblocks = SEQ / (WAVES * BM);          // 16
  const int bh = blockIdx.x / qblocks;             // head index (B*H = 64)
  const int qb = blockIdx.x % qblocks;
  const int q0 = qb * (WAVES * BM) + wave * BM;    // first q row of this wave

  // fold log2(e) into the softmax scale: scores land in log2 units
  const float scale = scale_p[0] * 1.44269504088896340736f;
  const float* __restrict__ Qh = Q + (size_t)bh * SEQ * DH;
  const float* __restrict__ Kh = K + (size_t)bh * SEQ * DH;
  const float* __restrict__ Vh = V + (size_t)bh * SEQ * DH;
  float* __restrict__       Oh = O + (size_t)bh * SEQ * DH;

  // staging assignment: each thread owns one quarter-row of the 64x64 block
  const int srow = tid & 63;        // kv row within block
  const int sq   = tid >> 6;        // quarter: d/kv-col range [16*sq, 16*sq+15]

  // ---- load Q tile (16x64) as two bf16 A operands, scale pre-folded ------
  // A 16x32 layout: lane m = lane%16; element e -> k = (e%8) + 16*(e/8) + 8h
  bf16x16 aq[2];
  {
    const float* qrow = Qh + (size_t)(q0 + n) * DH;
    #pragma unroll
    for (int p = 0; p < 2; ++p) {
      const float* c0 = qrow + p * 32 + 8 * h;        // k = 8h .. 8h+7
      const float* c1 = qrow + p * 32 + 16 + 8 * h;   // k = 16+8h .. 16+8h+7
      bf16x16 a;
      #pragma unroll
      for (int e = 0; e < 8; ++e) a[e]     = f2bf(c0[e] * scale);
      #pragma unroll
      for (int e = 0; e < 8; ++e) a[e + 8] = f2bf(c1[e] * scale);
      aq[p] = a;
    }
  }

  // all-ones B tile: P x ones accumulates the softmax row sums via WMMA
  bf16x16 onesb;
  #pragma unroll
  for (int e = 0; e < 16; ++e) onesb[e] = f2bf(1.0f);

  f32x8 acc[4] = {f32x8{}, f32x8{}, f32x8{}, f32x8{}};  // out cols, 4 tiles
  f32x8 accL = {};                                      // row sums (all cols equal)
  float mrow[8];
  #pragma unroll
  for (int r = 0; r < 8; ++r) mrow[r] = -1e30f;

  for (int kv0 = 0; kv0 < SEQ; kv0 += BN) {
    // ---- cooperative staging of K (row-major) and V (transposed) --------
    __syncthreads();   // all waves done reading previous K/V tiles
    {
      const float* gk = Kh + (size_t)(kv0 + srow) * DH + 16 * sq;
      bf16x8 k0, k1;
      #pragma unroll
      for (int j = 0; j < 8; ++j) { k0[j] = f2bf(gk[j]); k1[j] = f2bf(gk[8 + j]); }
      *(bf16x8*)&LdsK[srow][16 * sq]     = k0;
      *(bf16x8*)&LdsK[srow][16 * sq + 8] = k1;

      const float* gv = Vh + (size_t)(kv0 + srow) * DH + 16 * sq;
      #pragma unroll
      for (int j = 0; j < 16; ++j) LdsVt[16 * sq + j][srow] = f2bf(gv[j]);
    }
    if (kv0 + BN < SEQ) {   // speculative prefetch of next K block
      __builtin_prefetch(Kh + (size_t)(kv0 + BN + srow) * DH, 0, 3);
    }
    __syncthreads();   // staged tiles visible to all waves

    // ---- S = (Q*scale) K^T : four 16x16 f32 tiles ------------------------
    // B 32x16 layout: lane col = lane%16 (kv); element e -> d = p*32 + (e%8) + 16*(e/8) + 8h
    f32x8 s[4] = {f32x8{}, f32x8{}, f32x8{}, f32x8{}};
    #pragma unroll
    for (int p = 0; p < 2; ++p) {
      #pragma unroll
      for (int t = 0; t < 4; ++t) {
        union { bf16x16 v; bf16x8 half[2]; } kb;
        kb.half[0] = *(const bf16x8*)&LdsK[t * 16 + n][p * 32 + 8 * h];
        kb.half[1] = *(const bf16x8*)&LdsK[t * 16 + n][p * 32 + 16 + 8 * h];
        s[t] = __builtin_amdgcn_wmma_f32_16x16x32_bf16(false, aq[p], false, kb.v,
                                                       (short)0, s[t], false, false);
      }
    }

    // ---- online softmax (C layout: row m = r + 8h spans 16 lanes) --------
    float alpha[8];
    #pragma unroll
    for (int r = 0; r < 8; ++r) {
      float cm = fmaxf(fmaxf(s[0][r], s[1][r]), fmaxf(s[2][r], s[3][r]));
      cm = rmax16(cm);
      float mn = fmaxf(mrow[r], cm);
      alpha[r] = fast_exp2(mrow[r] - mn);
      mrow[r]  = mn;
      const int m = r + 8 * h;
      #pragma unroll
      for (int t = 0; t < 4; ++t) {
        Pl[wave][m][t * 16 + n] = f2bf(fast_exp2(s[t][r] - mn));
      }
    }
    #pragma unroll
    for (int r = 0; r < 8; ++r) {
      acc[0][r] *= alpha[r]; acc[1][r] *= alpha[r];
      acc[2][r] *= alpha[r]; acc[3][r] *= alpha[r];
      accL[r]   *= alpha[r];
    }

    __builtin_amdgcn_wave_barrier();   // order wave-private LDS stores/loads

    // ---- reload P in A layout (16x64 -> two 16x32 operands) --------------
    bf16x16 pa0, pa1;
    {
      union { bf16x16 v; bf16x8 half[2]; } pu;
      pu.half[0] = *(const bf16x8*)&Pl[wave][n][8 * h];
      pu.half[1] = *(const bf16x8*)&Pl[wave][n][16 + 8 * h];
      pa0 = pu.v;
      pu.half[0] = *(const bf16x8*)&Pl[wave][n][32 + 8 * h];
      pu.half[1] = *(const bf16x8*)&Pl[wave][n][48 + 8 * h];
      pa1 = pu.v;
    }
    __builtin_amdgcn_wave_barrier();

    // ---- row sums: accL += P * ones (two WMMAs, no butterfly) ------------
    accL = __builtin_amdgcn_wmma_f32_16x16x32_bf16(false, pa0, false, onesb,
                                                   (short)0, accL, false, false);
    accL = __builtin_amdgcn_wmma_f32_16x16x32_bf16(false, pa1, false, onesb,
                                                   (short)0, accL, false, false);

    // ---- O += P V : B operands from transposed Vt, contiguous ds reads ---
    // B 32x16 layout: lane col = lane%16 (d); element e -> kv = c*32 + (e%8) + 16*(e/8) + 8h
    #pragma unroll
    for (int t = 0; t < 4; ++t) {
      union { bf16x16 v; bf16x8 half[2]; } vb0, vb1;
      vb0.half[0] = *(const bf16x8*)&LdsVt[t * 16 + n][8 * h];
      vb0.half[1] = *(const bf16x8*)&LdsVt[t * 16 + n][16 + 8 * h];
      vb1.half[0] = *(const bf16x8*)&LdsVt[t * 16 + n][32 + 8 * h];
      vb1.half[1] = *(const bf16x8*)&LdsVt[t * 16 + n][48 + 8 * h];
      acc[t] = __builtin_amdgcn_wmma_f32_16x16x32_bf16(false, pa0, false, vb0.v,
                                                       (short)0, acc[t], false, false);
      acc[t] = __builtin_amdgcn_wmma_f32_16x16x32_bf16(false, pa1, false, vb1.v,
                                                       (short)0, acc[t], false, false);
    }
  }

  // ---- normalize and store (C layout: col n, rows r + 8h) ----------------
  #pragma unroll
  for (int r = 0; r < 8; ++r) {
    const float inv = fast_rcp(accL[r]);
    const size_t row = (size_t)(q0 + r + 8 * h) * DH;
    Oh[row +  0 + n] = acc[0][r] * inv;
    Oh[row + 16 + n] = acc[1][r] * inv;
    Oh[row + 32 + n] = acc[2][r] * inv;
    Oh[row + 48 + n] = acc[3][r] * inv;
  }
}

extern "C" void kernel_launch(void* const* d_in, const int* in_sizes, int n_in,
                              void* d_out, int out_size, void* d_ws, size_t ws_size,
                              hipStream_t stream) {
  (void)in_sizes; (void)n_in; (void)out_size; (void)d_ws; (void)ws_size;
  const float* Q = (const float*)d_in[0];
  const float* K = (const float*)d_in[1];
  const float* V = (const float*)d_in[2];
  // d_in[3] = is_causal: no-op in the reference math, intentionally unused.
  const float* scale = (const float*)d_in[4];
  float* O = (float*)d_out;

  const int BH = 4 * 16;                       // B*H heads
  const int qblocks = SEQ / (WAVES * BM);      // 16 q-blocks per head
  dim3 grid(BH * qblocks);                     // 1024 workgroups
  dim3 block(WAVES * 32);                      // 256 threads = 8 wave32

  hipLaunchKernelGGL(fa_fwd_bf16wmma, grid, block, 0, stream, Q, K, V, scale, O);
}
